// MergedLinearFormer_17463337026246
// MI455X (gfx1250) — matrix operand order repaired and checked by reading, not compiled, not run on hardware
//
#include <hip/hip_runtime.h>

#define BATCH 8
#define SEQ   2048
#define DIM   512
#define TQ    16
#define SCALE 0.044194173824159216f /* 1/sqrt(512) */

#if __has_builtin(__builtin_amdgcn_sched_barrier)
#define SCHED_FENCE() __builtin_amdgcn_sched_barrier(0)
#else
#define SCHED_FENCE()
#endif

typedef __attribute__((ext_vector_type(16))) _Float16 v16h;
typedef __attribute__((ext_vector_type(8)))  _Float16 v8h;
typedef __attribute__((ext_vector_type(8)))  float    v8f;

__device__ __forceinline__ v16h frag_cat(v8h lo, v8h hi) {
  return __builtin_shufflevector(lo, hi, 0,1,2,3,4,5,6,7,8,9,10,11,12,13,14,15);
}

// A fragment: 16x32 f16 tile, row-major with leading dim ld (elements).
// ISA 7.12.2: lanes 0-15 row=lane K={0..7,16..23}; lanes 16-31 row=lane-16 K={8..15,24..31}
__device__ __forceinline__ v16h load_frag_a(const _Float16* src, int ld, int lane) {
  int row  = lane & 15;
  int koff = (lane & 16) ? 8 : 0;
  const _Float16* p = src + (size_t)row * ld + koff;
  v8h lo = *(const v8h*)(p);
  v8h hi = *(const v8h*)(p + 16);
  return frag_cat(lo, hi);
}

// B fragment: 32x16 f16 tile supplied as its transpose [N][K] row-major (ld elems).
// lanes 0-15 hold K={0..7,16..23}, lanes 16-31 hold K={8..15,24..31}, N striped (matches A).
__device__ __forceinline__ v16h load_frag_b(const _Float16* srcT, int ld, int lane) {
  int n     = lane & 15;
  int kbase = (lane & 16) ? 8 : 0;
  const _Float16* p = srcT + (size_t)n * ld + kbase;
  v8h lo = *(const v8h*)(p);
  v8h hi = *(const v8h*)(p + 16);
  return frag_cat(lo, hi);
}

__device__ __forceinline__ v8f wmma16(v16h a, v16h b, v8f c) {
  return __builtin_amdgcn_wmma_f32_16x16x32_f16(false, a, false, b, (short)0, c, false, false);
}

// ---------------- Pre-pass 1: x -> f16 row-major (xh) and f16 transposed (xt) --------
__global__ void k_cvt_x(const float* __restrict__ x,
                        _Float16* __restrict__ xh,
                        _Float16* __restrict__ xt) {
  __shared__ _Float16 tile[32][33];
  int b  = blockIdx.z;
  int t0 = blockIdx.x * 32;
  int d0 = blockIdx.y * 32;
  int tx = threadIdx.x, ty = threadIdx.y; // 32 x 8
#pragma unroll
  for (int j = 0; j < 4; ++j) {
    int t = t0 + ty + j * 8;
    float v = x[((size_t)b * SEQ + t) * DIM + d0 + tx];
    _Float16 h = (_Float16)v;
    tile[ty + j * 8][tx] = h;
    xh[((size_t)b * SEQ + t) * DIM + d0 + tx] = h;
  }
  __syncthreads();
#pragma unroll
  for (int j = 0; j < 4; ++j) {
    int d = d0 + ty + j * 8;
    xt[((size_t)b * DIM + d) * SEQ + t0 + tx] = tile[tx][ty + j * 8];
  }
}

// ---------------- Pre-pass 2: QK^T and VO^T in f16 ----------------------------------
__global__ void k_cvt_w(const float* __restrict__ QK, const float* __restrict__ VO,
                        _Float16* __restrict__ qkt, _Float16* __restrict__ vot) {
  int i = blockIdx.x * 256 + threadIdx.x; // over DIM*DIM
  int e = i % DIM;
  int d = i / DIM;
  qkt[(size_t)e * DIM + d] = (_Float16)QK[i];
  vot[(size_t)e * DIM + d] = (_Float16)VO[i];
}

// ---------------- Kernel 3: q = (x @ QK) * scale, f16 out ---------------------------
// 8 waves; each wave owns 4 N-tiles. Software-pipelined (2-deep) over K.
__launch_bounds__(256)
__global__ void k_qproj(const _Float16* __restrict__ xh,
                        const _Float16* __restrict__ qkt,
                        _Float16* __restrict__ qh) {
  int wg   = blockIdx.x;           // one 16-row tile of (B*T)
  int wave = threadIdx.x >> 5;     // 0..7
  int lane = threadIdx.x & 31;
  const _Float16* abase = xh + (size_t)wg * TQ * DIM;
  const _Float16* bbase = qkt + (size_t)(wave * 64) * DIM;

  v8f acc[4];
#pragma unroll
  for (int j = 0; j < 4; ++j) acc[j] = (v8f){};

  v16h af0, af1, vb0[4], vb1[4];
  af0 = load_frag_a(abase, DIM, lane);
#pragma unroll
  for (int j = 0; j < 4; ++j) vb0[j] = load_frag_b(bbase + (size_t)(j * 16) * DIM, DIM, lane);
  SCHED_FENCE();
#pragma unroll
  for (int k = 0; k < 16; k += 2) {
    af1 = load_frag_a(abase + (k + 1) * 32, DIM, lane);
#pragma unroll
    for (int j = 0; j < 4; ++j)
      vb1[j] = load_frag_b(bbase + (size_t)(j * 16) * DIM + (k + 1) * 32, DIM, lane);
    SCHED_FENCE();
#pragma unroll
    for (int j = 0; j < 4; ++j) acc[j] = wmma16(af0, vb0[j], acc[j]);
    SCHED_FENCE();
    if (k + 2 < 16) {
      af0 = load_frag_a(abase + (k + 2) * 32, DIM, lane);
#pragma unroll
      for (int j = 0; j < 4; ++j)
        vb0[j] = load_frag_b(bbase + (size_t)(j * 16) * DIM + (k + 2) * 32, DIM, lane);
    }
    SCHED_FENCE();
#pragma unroll
    for (int j = 0; j < 4; ++j) acc[j] = wmma16(af1, vb1[j], acc[j]);
    SCHED_FENCE();
  }

  int col16 = lane & 15;
  int rbase = (lane & 16) ? 8 : 0;
#pragma unroll
  for (int j = 0; j < 4; ++j) {
    int col = wave * 64 + j * 16 + col16;
#pragma unroll
    for (int r = 0; r < 8; ++r)
      qh[((size_t)wg * TQ + rbase + r) * DIM + col] = (_Float16)(acc[j][r] * SCALE);
  }
}

// ---------------- Kernel 4: attention + output projection ---------------------------
// 512 threads (16 waves). Scores in registers; exact softmax; pipelined GEMMs.
__launch_bounds__(512)
__global__ void k_attn(const _Float16* __restrict__ qh,   // [B*T][D] scaled q
                       const _Float16* __restrict__ xh,   // [B*T][D] keys (as [u][d])
                       const _Float16* __restrict__ xt,   // [B][D][T] values transposed
                       const _Float16* __restrict__ vot,  // [D][D] VO^T
                       float* __restrict__ out) {
  __shared__ _Float16 P[TQ][SEQ];     // 64 KB probs (f16); reused as ctx after GEMM2
  __shared__ float redm[16][16];
  __shared__ float reds[16][16];

  int wg   = blockIdx.x;
  int b    = wg / (SEQ / TQ);
  int t0   = (wg % (SEQ / TQ)) * TQ;
  int tid  = threadIdx.x;
  int wave = tid >> 5;                 // 0..15
  int lane = tid & 31;
  int col16 = lane & 15;
  int rbase = (lane & 16) ? 8 : 0;

  const _Float16* qbase  = qh + (size_t)(b * SEQ + t0) * DIM;
  const _Float16* xtbase = xt + (size_t)b * DIM * SEQ;
  // wave owns 128 score columns = 8 tiles
  const _Float16* kvbase = xh + (size_t)b * SEQ * DIM + (size_t)(wave * 128) * DIM;

  // ---- GEMM1: S = q @ x^T, 2-deep pipelined over K ----
  v8f sacc[8];
#pragma unroll
  for (int j = 0; j < 8; ++j) sacc[j] = (v8f){};

  {
    v16h af0, af1, vb0[8], vb1[8];
    af0 = load_frag_a(qbase, DIM, lane);
#pragma unroll
    for (int j = 0; j < 8; ++j) vb0[j] = load_frag_b(kvbase + (size_t)(j * 16) * DIM, DIM, lane);
    SCHED_FENCE();
#pragma unroll
    for (int k = 0; k < 16; k += 2) {
      af1 = load_frag_a(qbase + (k + 1) * 32, DIM, lane);
#pragma unroll
      for (int j = 0; j < 8; ++j)
        vb1[j] = load_frag_b(kvbase + (size_t)(j * 16) * DIM + (k + 1) * 32, DIM, lane);
      SCHED_FENCE();
#pragma unroll
      for (int j = 0; j < 8; ++j) sacc[j] = wmma16(af0, vb0[j], sacc[j]);
      SCHED_FENCE();
      if (k + 2 < 16) {
        af0 = load_frag_a(qbase + (k + 2) * 32, DIM, lane);
#pragma unroll
        for (int j = 0; j < 8; ++j)
          vb0[j] = load_frag_b(kvbase + (size_t)(j * 16) * DIM + (k + 2) * 32, DIM, lane);
      }
      SCHED_FENCE();
#pragma unroll
      for (int j = 0; j < 8; ++j) sacc[j] = wmma16(af1, vb1[j], sacc[j]);
      SCHED_FENCE();
    }
  }

  // ---- softmax (exact): register partials -> shfl within 16-lane half -> LDS combine
  float pm[8];
#pragma unroll
  for (int r = 0; r < 8; ++r) {
    float m = sacc[0][r];
#pragma unroll
    for (int j = 1; j < 8; ++j) m = fmaxf(m, sacc[j][r]);
    pm[r] = m;
  }
#pragma unroll
  for (int msk = 1; msk < 16; msk <<= 1)
#pragma unroll
    for (int r = 0; r < 8; ++r) pm[r] = fmaxf(pm[r], __shfl_xor(pm[r], msk, 32));
  if (lane == 0)
#pragma unroll
    for (int r = 0; r < 8; ++r) redm[wave][r] = pm[r];
  if (lane == 16)
#pragma unroll
    for (int r = 0; r < 8; ++r) redm[wave][8 + r] = pm[r];
  __syncthreads();

  float gm[8];
#pragma unroll
  for (int r = 0; r < 8; ++r) {
    float m = redm[0][rbase + r];
#pragma unroll
    for (int w = 1; w < 16; ++w) m = fmaxf(m, redm[w][rbase + r]);
    gm[r] = m;
  }

  float ps[8];
#pragma unroll
  for (int r = 0; r < 8; ++r) ps[r] = 0.f;
#pragma unroll
  for (int j = 0; j < 8; ++j)
#pragma unroll
    for (int r = 0; r < 8; ++r) {
      float e = __expf(sacc[j][r] - gm[r]);
      sacc[j][r] = e;
      ps[r] += e;
    }
#pragma unroll
  for (int msk = 1; msk < 16; msk <<= 1)
#pragma unroll
    for (int r = 0; r < 8; ++r) ps[r] += __shfl_xor(ps[r], msk, 32);
  if (lane == 0)
#pragma unroll
    for (int r = 0; r < 8; ++r) reds[wave][r] = ps[r];
  if (lane == 16)
#pragma unroll
    for (int r = 0; r < 8; ++r) reds[wave][8 + r] = ps[r];
  __syncthreads();

  float inv[8];
#pragma unroll
  for (int r = 0; r < 8; ++r) {
    float s = 0.f;
#pragma unroll
    for (int w = 0; w < 16; ++w) s += reds[w][rbase + r];
    inv[r] = 1.f / s;
  }

  // normalized probs -> LDS in row-major f16 (A-fragment friendly)
#pragma unroll
  for (int j = 0; j < 8; ++j) {
    int col = wave * 128 + j * 16 + col16;
#pragma unroll
    for (int r = 0; r < 8; ++r)
      P[rbase + r][col] = (_Float16)(sacc[j][r] * inv[r]);
  }
  __syncthreads();

  // ---- GEMM2: ctx = P @ V. Each wave owns 32 output columns (2 tiles). Pipelined. ----
  int n0 = wave * 32;
  v8f cacc[2];
  cacc[0] = (v8f){}; cacc[1] = (v8f){};
  {
    v16h pa0, pa1, vb0[2], vb1[2];
    pa0 = load_frag_a(&P[0][0], SEQ, lane);
#pragma unroll
    for (int j = 0; j < 2; ++j) vb0[j] = load_frag_b(xtbase + (size_t)(n0 + j * 16) * SEQ, SEQ, lane);
    SCHED_FENCE();
    for (int kt = 0; kt < SEQ / 32; kt += 2) {
      pa1 = load_frag_a(&P[0][(kt + 1) * 32], SEQ, lane);
#pragma unroll
      for (int j = 0; j < 2; ++j)
        vb1[j] = load_frag_b(xtbase + (size_t)(n0 + j * 16) * SEQ + (kt + 1) * 32, SEQ, lane);
      SCHED_FENCE();
#pragma unroll
      for (int j = 0; j < 2; ++j) cacc[j] = wmma16(pa0, vb0[j], cacc[j]);
      SCHED_FENCE();
      if (kt + 2 < SEQ / 32) {
        pa0 = load_frag_a(&P[0][(kt + 2) * 32], SEQ, lane);
#pragma unroll
        for (int j = 0; j < 2; ++j)
          vb0[j] = load_frag_b(xtbase + (size_t)(n0 + j * 16) * SEQ + (kt + 2) * 32, SEQ, lane);
      }
      SCHED_FENCE();
#pragma unroll
      for (int j = 0; j < 2; ++j) cacc[j] = wmma16(pa1, vb1[j], cacc[j]);
      SCHED_FENCE();
    }
  }
  __syncthreads();   // all waves done reading P

  _Float16* CT = &P[0][0];  // reuse P storage: ctx f16 [TQ][DIM]
#pragma unroll
  for (int j = 0; j < 2; ++j) {
    int col = n0 + j * 16 + col16;
#pragma unroll
    for (int r = 0; r < 8; ++r)
      CT[(size_t)(rbase + r) * DIM + col] = (_Float16)cacc[j][r];
  }
  __syncthreads();

  // ---- GEMM3: out = ctx @ VO. Pipelined. ----
  v8f oacc[2];
  oacc[0] = (v8f){}; oacc[1] = (v8f){};
  {
    v16h ca0, ca1, vb0[2], vb1[2];
    ca0 = load_frag_a(CT, DIM, lane);
#pragma unroll
    for (int j = 0; j < 2; ++j) vb0[j] = load_frag_b(vot + (size_t)(n0 + j * 16) * DIM, DIM, lane);
    SCHED_FENCE();
#pragma unroll
    for (int kt = 0; kt < DIM / 32; kt += 2) {
      ca1 = load_frag_a(CT + (kt + 1) * 32, DIM, lane);
#pragma unroll
      for (int j = 0; j < 2; ++j)
        vb1[j] = load_frag_b(vot + (size_t)(n0 + j * 16) * DIM + (kt + 1) * 32, DIM, lane);
      SCHED_FENCE();
#pragma unroll
      for (int j = 0; j < 2; ++j) oacc[j] = wmma16(ca0, vb0[j], oacc[j]);
      SCHED_FENCE();
      if (kt + 2 < DIM / 32) {
        ca0 = load_frag_a(CT + (kt + 2) * 32, DIM, lane);
#pragma unroll
        for (int j = 0; j < 2; ++j)
          vb0[j] = load_frag_b(vot + (size_t)(n0 + j * 16) * DIM + (kt + 2) * 32, DIM, lane);
      }
      SCHED_FENCE();
#pragma unroll
      for (int j = 0; j < 2; ++j) oacc[j] = wmma16(ca1, vb1[j], oacc[j]);
      SCHED_FENCE();
    }
  }
  float* obase = out + (size_t)(b * SEQ + t0) * DIM;
#pragma unroll
  for (int j = 0; j < 2; ++j) {
    int col = n0 + j * 16 + col16;
#pragma unroll
    for (int r = 0; r < 8; ++r)
      obase[(size_t)(rbase + r) * DIM + col] = oacc[j][r];
  }
}

extern "C" void kernel_launch(void* const* d_in, const int* in_sizes, int n_in,
                              void* d_out, int out_size, void* d_ws, size_t ws_size,
                              hipStream_t stream) {
  const float* x  = (const float*)d_in[0];
  const float* QK = (const float*)d_in[1];
  const float* VO = (const float*)d_in[2];
  float* out = (float*)d_out;

  const size_t XB = (size_t)BATCH * SEQ * DIM * sizeof(_Float16); // 16 MB
  const size_t WB = (size_t)DIM * DIM * sizeof(_Float16);         // 512 KB
  if (ws_size < 3 * XB + 2 * WB) return; // need ~49 MB scratch

  char* ws = (char*)d_ws;
  _Float16* xh  = (_Float16*)(ws);
  _Float16* xt  = (_Float16*)(ws + XB);
  _Float16* qh  = (_Float16*)(ws + 2 * XB);
  _Float16* qkt = (_Float16*)(ws + 3 * XB);
  _Float16* vot = (_Float16*)(ws + 3 * XB + WB);

  k_cvt_x<<<dim3(SEQ / 32, DIM / 32, BATCH), dim3(32, 8), 0, stream>>>(x, xh, xt);
  k_cvt_w<<<(DIM * DIM) / 256, 256, 0, stream>>>(QK, VO, qkt, vot);
  k_qproj<<<(BATCH * SEQ) / TQ, 256, 0, stream>>>(xh, qkt, qh);
  k_attn<<<(BATCH * SEQ) / TQ, 512, 0, stream>>>(qh, xh, xt, vot, out);
}
